// ConvNext_mix_feat_47880295416525
// MI455X (gfx1250) — matrix operand order repaired and checked by reading, not compiled, not run on hardware
//
#include <hip/hip_runtime.h>
#include <hip/hip_bf16.h>
#include <math.h>

// ---------------------------------------------------------------------------
// Sizes (fixed by the reference)
// ---------------------------------------------------------------------------
#define B_    32
#define C_    384
#define RHO_  25
#define WP_   256
#define HC_   64
#define WC_   64
#define D_    768           // 2*C
#define NH_   40            // hidden units
#define NPAD_ 48            // hidden padded to 3 WMMA N-tiles
#define KD_   768           // GEMM K

typedef _Float16 v16h __attribute__((ext_vector_type(16)));
typedef _Float16 v8h  __attribute__((ext_vector_type(8)));
typedef float    v8f  __attribute__((ext_vector_type(8)));

// ---------------------------------------------------------------------------
// Kernel A: fused bilinear grid-sample + width-mean.
// One workgroup per (b, rho). 256 threads = 8 waves.
// Phase 1: each thread computes the 4 bilinear (index, weight) pairs for its
//          wp column once -> LDS.
// Phase 2: each lane caches its 8 wp-positions' params in REGISTERS (they are
//          invariant across the channel loop), then wave w handles channels
//          c = w, w+8, ...; 32 lanes split the 256 wp positions, wave-shuffle
//          reduce. Polar means use per-lane contiguous float4 loads (b128).
// Output: feats[rho][b][c] (c<384 polar mean, c>=384 sampled-cart mean).
// ---------------------------------------------------------------------------
__global__ __launch_bounds__(256)
void sample_mean_kernel(const float* __restrict__ polar,
                        const float* __restrict__ cart,
                        const float* __restrict__ grid,
                        float* __restrict__ feats)
{
    __shared__ int   s_idx[4][WP_];
    __shared__ float s_w[4][WP_];

    const int blk = blockIdx.x;            // 0 .. B_*RHO_-1
    const int b   = blk / RHO_;
    const int rho = blk % RHO_;
    const int tid = threadIdx.x;

    // ---- Phase 1: sampling parameters for wp = tid -------------------------
    {
        const int wp = tid;
        const float gx = grid[(((b * RHO_ + rho) * WP_ + wp) * 2) + 0];
        const float gy = grid[(((b * RHO_ + rho) * WP_ + wp) * 2) + 1];
        const float ix = (gx + 1.0f) * (0.5f * WC_) - 0.5f;
        const float iy = (gy + 1.0f) * (0.5f * HC_) - 0.5f;
        const float ix0f = floorf(ix);
        const float iy0f = floorf(iy);
        const float tx = ix - ix0f;
        const float ty = iy - iy0f;
        const int ix0 = (int)ix0f;
        const int iy0 = (int)iy0f;

        const int   cx[4] = { ix0, ix0 + 1, ix0,     ix0 + 1 };
        const int   cy[4] = { iy0, iy0,     iy0 + 1, iy0 + 1 };
        const float cw[4] = { (1.f - tx) * (1.f - ty), tx * (1.f - ty),
                              (1.f - tx) * ty,         tx * ty };
        #pragma unroll
        for (int k = 0; k < 4; ++k) {
            const bool valid = (cx[k] >= 0) && (cx[k] < WC_) &&
                               (cy[k] >= 0) && (cy[k] < HC_);
            s_idx[k][wp] = valid ? (cy[k] * WC_ + cx[k]) : 0;
            s_w[k][wp]   = valid ? cw[k] : 0.0f;
        }
    }
    __syncthreads();

    const int wave = tid >> 5;
    const int lane = tid & 31;
    const float inv_wp = 1.0f / (float)WP_;

    // ---- Cache this lane's sampling params in registers (loop-invariant) ---
    int   ridx[4][WP_ / 32];
    float rw[4][WP_ / 32];
    #pragma unroll
    for (int j = 0; j < WP_ / 32; ++j) {
        const int wp = lane + 32 * j;
        #pragma unroll
        for (int k = 0; k < 4; ++k) {
            ridx[k][j] = s_idx[k][wp];
            rw[k][j]   = s_w[k][wp];
        }
    }

    // ---- Phase 2a: polar mean (b128 loads, coalesced across the wave) ------
    for (int c = wave; c < C_; c += 8) {
        const float* p = polar + (((size_t)(b * C_ + c) * RHO_ + rho) * WP_);
        const float4 f0 = *(const float4*)(p + lane * 8);
        const float4 f1 = *(const float4*)(p + lane * 8 + 4);
        float s = ((f0.x + f0.y) + (f0.z + f0.w)) +
                  ((f1.x + f1.y) + (f1.z + f1.w));
        #pragma unroll
        for (int off = 16; off >= 1; off >>= 1) s += __shfl_xor(s, off, 32);
        if (lane == 0)
            feats[((size_t)(rho * B_ + b)) * D_ + c] = s * inv_wp;
    }

    // ---- Phase 2b: bilinear-sampled cart mean (gathers, L2-resident) -------
    for (int c = wave; c < C_; c += 8) {
        const float* cp = cart + ((size_t)(b * C_ + c)) * (HC_ * WC_);
        float s = 0.0f;
        #pragma unroll
        for (int j = 0; j < WP_ / 32; ++j) {
            s += rw[0][j] * cp[ridx[0][j]];
            s += rw[1][j] * cp[ridx[1][j]];
            s += rw[2][j] * cp[ridx[2][j]];
            s += rw[3][j] * cp[ridx[3][j]];
        }
        #pragma unroll
        for (int off = 16; off >= 1; off >>= 1) s += __shfl_xor(s, off, 32);
        if (lane == 0)
            feats[((size_t)(rho * B_ + b)) * D_ + C_ + c] = s * inv_wp;
    }
}

// ---------------------------------------------------------------------------
// Prep: pack W1 weights (first 768 K rows) into f16, N-major + zero-padded:
// Wt[h][n][k], h in [0,25), n in [0,48) (n>=40 -> 0), k in [0,768).
// ---------------------------------------------------------------------------
__global__ __launch_bounds__(256)
void prep_weights_kernel(const float* __restrict__ W1_0,
                         const float* __restrict__ W1s,
                         _Float16* __restrict__ Wt)
{
    const int idx = blockIdx.x * 256 + threadIdx.x;
    const int total = RHO_ * NPAD_ * KD_;
    if (idx >= total) return;
    const int k = idx % KD_;
    const int t = idx / KD_;
    const int n = t % NPAD_;
    const int h = t / NPAD_;
    float v = 0.0f;
    if (n < NH_) {
        v = (h == 0) ? W1_0[k * NH_ + n]
                     : W1s[(((h - 1) * (D_ + 1)) + k) * NH_ + n];
    }
    Wt[idx] = (_Float16)v;
}

// ---------------------------------------------------------------------------
// Kernel B: per-head GEMM  pre[h] = feats[h] (32x768) @ W1_h (768x40)
// One wave (32 lanes) per head. f16 inputs, f32 accumulate via
// v_wmma_f32_16x16x32_f16. 2 M-tiles x 3 N-tiles x 24 K-steps = 144 WMMA.
//
// Fragment layouts (ISA 7.12.2, wave32):
//  A (16-bit, 16x32): lanes 0-15: M=lane,    elems 0-7 = K k0+0..7,
//                                            elems 8-15 = K k0+16..23
//                     lanes 16-31: M=lane-16, K offsets +8 (8..15 / 24..31)
//  B (16-bit, 32x16): lanes 0-15: N=lane,    elems 0-7 = K k0+0..7, elems 8-15 = K k0+16..23
//                     lanes 16-31: N=lane-16, K offsets +8 (8..15 / 24..31)
//                     (same K order as A)
//  C/D (f32 16x16):   N = lane&15, M = vgpr + 8*(lane>=16)
// ---------------------------------------------------------------------------
__global__ __launch_bounds__(32)
void head_gemm_wmma_kernel(const float* __restrict__ feats,
                           const _Float16* __restrict__ Wt,
                           float* __restrict__ pre)
{
    const int h    = blockIdx.x;        // head 0..24
    const int lane = threadIdx.x;
    const int half = lane >> 4;
    const int lrow = lane & 15;
    const int akoff = half * 8;         // A-fragment K sub-offset
    const int bkoff = half * 8;         // B-fragment K sub-offset (same as A)

    const float*    Abase = feats + (size_t)h * B_ * D_;
    const _Float16* Bbase = Wt    + (size_t)h * NPAD_ * KD_;

    v8f acc[2][3];
    #pragma unroll
    for (int mt = 0; mt < 2; ++mt)
        #pragma unroll
        for (int nt = 0; nt < 3; ++nt)
            #pragma unroll
            for (int e = 0; e < 8; ++e) acc[mt][nt][e] = 0.0f;

    for (int k0 = 0; k0 < KD_; k0 += 32) {
        v16h a[2];
        #pragma unroll
        for (int mt = 0; mt < 2; ++mt) {
            const float* p = Abase + (size_t)(lrow + 16 * mt) * D_ + k0 + akoff;
            #pragma unroll
            for (int e = 0; e < 8; ++e) a[mt][e]     = (_Float16)p[e];
            #pragma unroll
            for (int e = 0; e < 8; ++e) a[mt][8 + e] = (_Float16)p[16 + e];
        }
        v16h bf[3];
        #pragma unroll
        for (int nt = 0; nt < 3; ++nt) {
            const _Float16* q = Bbase + (size_t)(lrow + 16 * nt) * KD_ + k0 + bkoff;
            const v8h lo = *(const v8h*)(q);      // K = bkoff+0..7
            const v8h hi = *(const v8h*)(q + 16); // K = bkoff+16..23
            #pragma unroll
            for (int e = 0; e < 8; ++e) { bf[nt][e] = lo[e]; bf[nt][8 + e] = hi[e]; }
        }
        #pragma unroll
        for (int mt = 0; mt < 2; ++mt)
            #pragma unroll
            for (int nt = 0; nt < 3; ++nt)
                acc[mt][nt] = __builtin_amdgcn_wmma_f32_16x16x32_f16(
                    false, a[mt], false, bf[nt], (short)0, acc[mt][nt],
                    false, false);
    }

    float* pr = pre + (size_t)h * B_ * NPAD_;
    #pragma unroll
    for (int mt = 0; mt < 2; ++mt)
        #pragma unroll
        for (int nt = 0; nt < 3; ++nt)
            #pragma unroll
            for (int r = 0; r < 8; ++r) {
                const int m = 16 * mt + 8 * half + r;   // batch row
                const int n = 16 * nt + lrow;           // hidden col
                pr[m * NPAD_ + n] = acc[mt][nt][r];
            }
}

// ---------------------------------------------------------------------------
// Kernel C: sequential head scan. One wave; lane = batch index.
// Recurrent scalar only multiplies W1 row 768, so each step is 40 GELU+MACs.
// ---------------------------------------------------------------------------
__device__ __forceinline__ float gelu_exact(float x) {
    return 0.5f * x * (1.0f + erff(x * 0.70710678118654752440f));
}

__global__ __launch_bounds__(32)
void head_scan_kernel(const float* __restrict__ pre,
                      const float* __restrict__ b1_0,
                      const float* __restrict__ W2_0,
                      const float* __restrict__ b2_0,
                      const float* __restrict__ W1s,
                      const float* __restrict__ b1s,
                      const float* __restrict__ W2s,
                      const float* __restrict__ b2s,
                      float* __restrict__ out)
{
    const int b = threadIdx.x;            // 0..31
    const float PI_F = 3.14159265358979323846f;

    // Head 0
    float o;
    {
        float s = 0.0f;
        #pragma unroll
        for (int j = 0; j < NH_; ++j) {
            const float x = pre[b * NPAD_ + j] + b1_0[j];
            s += gelu_exact(x) * W2_0[j];
        }
        o = s + b2_0[0];
        out[b * RHO_ + 0] = fminf(fmaxf(o, 0.0f), PI_F);
    }
    // Heads 1..24 (sequential carry on pre-clip o)
    for (int i = 1; i < RHO_; ++i) {
        const float* w1last = W1s + ((size_t)(i - 1) * (D_ + 1) + D_) * NH_;
        const float* bb = b1s + (i - 1) * NH_;
        const float* w2 = W2s + (i - 1) * NH_;
        float s = 0.0f;
        #pragma unroll
        for (int j = 0; j < NH_; ++j) {
            const float x = pre[(i * B_ + b) * NPAD_ + j] + bb[j] + o * w1last[j];
            s += gelu_exact(x) * w2[j];
        }
        o = s + b2s[i - 1];
        out[b * RHO_ + i] = fminf(fmaxf(o, 0.0f), PI_F);
    }
}

// ---------------------------------------------------------------------------
// Launch
// ---------------------------------------------------------------------------
extern "C" void kernel_launch(void* const* d_in, const int* in_sizes, int n_in,
                              void* d_out, int out_size, void* d_ws, size_t ws_size,
                              hipStream_t stream) {
    const float* polar = (const float*)d_in[0];   // [32,384,25,256]
    const float* cart  = (const float*)d_in[1];   // [32,384,64,64]
    const float* grid  = (const float*)d_in[2];   // [32,25,256,2]
    const float* W1_0  = (const float*)d_in[3];   // [768,40]
    const float* b1_0  = (const float*)d_in[4];   // [40]
    const float* W2_0  = (const float*)d_in[5];   // [40,1]
    const float* b2_0  = (const float*)d_in[6];   // [1]
    const float* W1s   = (const float*)d_in[7];   // [24,769,40]
    const float* b1s   = (const float*)d_in[8];   // [24,40]
    const float* W2s   = (const float*)d_in[9];   // [24,40,1]
    const float* b2s   = (const float*)d_in[10];  // [24,1]
    float* out = (float*)d_out;                   // [32,25]

    // Workspace layout
    char* ws = (char*)d_ws;
    float*    feats = (float*)ws;                                    // 25*32*768 f32
    _Float16* Wt    = (_Float16*)(ws + (size_t)RHO_ * B_ * D_ * 4);  // 25*48*768 f16
    float*    pre   = (float*)(ws + (size_t)RHO_ * B_ * D_ * 4
                                  + (size_t)RHO_ * NPAD_ * KD_ * 2); // 25*32*48 f32

    // 1) weight prep
    {
        const int total = RHO_ * NPAD_ * KD_;
        prep_weights_kernel<<<(total + 255) / 256, 256, 0, stream>>>(W1_0, W1s, Wt);
    }
    // 2) fused grid-sample + width-mean
    sample_mean_kernel<<<B_ * RHO_, 256, 0, stream>>>(polar, cart, grid, feats);
    // 3) WMMA per-head GEMM (pre-activations, rows 0..767 of W1)
    head_gemm_wmma_kernel<<<RHO_, 32, 0, stream>>>(feats, Wt, pre);
    // 4) sequential head scan + clip
    head_scan_kernel<<<1, 32, 0, stream>>>(pre, b1_0, W2_0, b2_0,
                                           W1s, b1s, W2s, b2s, out);
}
